// MVN_DDI_4483945857416
// MI455X (gfx1250) — matrix operand order repaired and checked by reading, not compiled, not run on hardware
//
#include <hip/hip_runtime.h>
#include <hip/hip_bf16.h>

// ---------------------------------------------------------------------------
// MVN-DDI forward for MI455X (gfx1250, wave32, WMMA).
// - Dense GEMMs: v_wmma_f32_16x16x32_bf16, fp32 accumulate; weights pre-
//   transposed to bf16 [N][Kp] so B frags are contiguous global b128 loads.
// - Explicit address_space(1) casts (via clang ext_vector types) force
//   global_load/global_store instead of flat.
// - Q/K/V/Skip fused over one staged X tile; X tile double-buffered elsewhere.
// - TransformerConv attention fully fused per graph: q/k tiles + softmax +
//   aggregation all in LDS (ds_max_u32 / ds_add_f32 atomics).
// ---------------------------------------------------------------------------

typedef __bf16 v16bf __attribute__((ext_vector_type(16)));
typedef float  v8f   __attribute__((ext_vector_type(8)));
typedef unsigned int u32x4 __attribute__((ext_vector_type(4)));
typedef float        f32x4 __attribute__((ext_vector_type(4)));

enum : int { NN = 65536, EE = 131072, BB = 512, PP = 128, EPGn = 256,
             HIDn = 128, EFn = 64, FNODE = 70, FEDGE = 6, NBLKn = 4,
             FPn = 1024 };

__device__ __forceinline__ float elu1(float x) { return x > 0.f ? x : (__expf(x) - 1.f); }

// --- explicit global-address-space accessors (avoid flat_load/flat_store) ---
typedef const u32x4 __attribute__((address_space(1)))* gcu4_t;
typedef const f32x4 __attribute__((address_space(1)))* gcf4_t;
typedef float       __attribute__((address_space(1)))* gf_t;

__device__ __forceinline__ u32x4 gload_u4(const void* p) { return *(gcu4_t)p; }
__device__ __forceinline__ f32x4 gload_f4(const void* p) { return *(gcf4_t)p; }
__device__ __forceinline__ void  gstore_f(void* p, float v) { *(gf_t)p = v; }

// monotonic float<->uint mapping for atomic max on shared memory
__device__ __forceinline__ unsigned fenc(float f) {
  unsigned b = __float_as_uint(f);
  return b ^ ((unsigned)((int)b >> 31) | 0x80000000u);
}
__device__ __forceinline__ float fdec(unsigned u) {
  unsigned b = u ^ ((unsigned)((int)(~u) >> 31) | 0x80000000u);
  return __uint_as_float(b);
}

union Frag { u32x4 u[2]; v16bf v; };
union Pack8 { __bf16 h[8]; u32x4 u; };

__device__ __forceinline__ void pack8(Pack8& p, f32x4 a, f32x4 b) {
  p.h[0] = (__bf16)a[0]; p.h[1] = (__bf16)a[1]; p.h[2] = (__bf16)a[2]; p.h[3] = (__bf16)a[3];
  p.h[4] = (__bf16)b[0]; p.h[5] = (__bf16)b[1]; p.h[6] = (__bf16)b[2]; p.h[7] = (__bf16)b[3];
}

// ---------------------------------------------------------------------------
// Weight prep: W[b][K][N] fp32 -> Wt[b][N][Kp] bf16, zero-padded K tail.
// ---------------------------------------------------------------------------
__global__ void k_wt(const float* __restrict__ W, __bf16* __restrict__ Wt,
                     int K, int N, int Kp, int total) {
  int i = blockIdx.x * 256 + threadIdx.x;
  if (i >= total) return;
  int kk = i % Kp; int rem = i / Kp; int n = rem % N; int b = rem / N;
  float v = (kk < K) ? W[((size_t)b * K + kk) * N + n] : 0.f;
  Wt[i] = (__bf16)v;
}

// ---------------------------------------------------------------------------
// Generic GEMM: out[M,N] = act(X[M,K] @ W + bias). Wt bf16 [N][Kp].
// M % 128 == 0, N == NT*16. ACT: 0 none, 1 ELU, 2 ReLU. Double-buffered sA,
// one barrier per k-step.
// ---------------------------------------------------------------------------
template <int ACT, int NT>
__global__ __launch_bounds__(256) void gemm_wmma(
    const float* __restrict__ X, const __bf16* __restrict__ Wt,
    const float* __restrict__ bias, float* __restrict__ out,
    int M, int N, int K, int Kp) {
  __shared__ __align__(16) __bf16 sA[2][128][40];
  const int tid = threadIdx.x, lane = tid & 31, wave = tid >> 5;
  const int mBase = blockIdx.x * 128;
  const bool kvec = ((K & 3) == 0);

  v8f acc[NT];
#pragma unroll
  for (int i = 0; i < NT; ++i)
#pragma unroll
    for (int j = 0; j < 8; ++j) acc[i][j] = 0.f;

  int buf = 0;
  for (int k0 = 0; k0 < K; k0 += 32, buf ^= 1) {
    {  // stage X tile (128 x 32) as bf16
      int r = tid >> 1, c0 = (tid & 1) * 16;
      if (kvec && (k0 + 32 <= K)) {
        const float* xp = X + (size_t)(mBase + r) * K + k0 + c0;
        f32x4 f0 = gload_f4(xp), f1 = gload_f4(xp + 4);
        f32x4 f2 = gload_f4(xp + 8), f3 = gload_f4(xp + 12);
        Pack8 p0, p1; pack8(p0, f0, f1); pack8(p1, f2, f3);
        *(u32x4*)&sA[buf][r][c0] = p0.u;
        *(u32x4*)&sA[buf][r][c0 + 8] = p1.u;
      } else {
        const float* xp = X + (size_t)(mBase + r) * K;
#pragma unroll
        for (int i = 0; i < 16; ++i) {
          int kk = k0 + c0 + i;
          sA[buf][r][c0 + i] = (__bf16)((kk < K) ? xp[kk] : 0.f);
        }
      }
    }
    __syncthreads();

    Frag a;
    int ar = (wave << 4) + (lane & 15);
    int ak = (lane < 16) ? 0 : 8;
    a.u[0] = *(const u32x4*)&sA[buf][ar][ak];
    a.u[1] = *(const u32x4*)&sA[buf][ar][ak + 16];

#pragma unroll
    for (int nt = 0; nt < NT; ++nt) {
      const __bf16* wr = Wt + (size_t)((nt << 4) + (lane & 15)) * Kp + k0 +
                         ((lane < 16) ? 0 : 8);
      Frag b;
      b.u[0] = gload_u4(wr);
      b.u[1] = gload_u4(wr + 16);
      acc[nt] = __builtin_amdgcn_wmma_f32_16x16x32_bf16(
          false, a.v, false, b.v, (short)0, acc[nt], false, false);
    }
  }

  int rowTop = mBase + (wave << 4) + ((lane < 16) ? 0 : 8);
#pragma unroll
  for (int nt = 0; nt < NT; ++nt) {
    int col = (nt << 4) + (lane & 15);
    float bv = bias ? bias[col] : 0.f;
#pragma unroll
    for (int j = 0; j < 8; ++j) {
      float vv = acc[nt][j] + bv;
      if (ACT == 1) vv = elu1(vv);
      if (ACT == 2) vv = fmaxf(vv, 0.f);
      gstore_f(out + (size_t)(rowTop + j) * N + col, vv);
    }
  }
}

// ---------------------------------------------------------------------------
// Fused Q/K/V/Skip GEMM: K=N=128. One staged X tile feeds 4 weight matrices.
// Manual per-weight helper keeps pointers as direct args -> global loads.
// ---------------------------------------------------------------------------
__device__ __forceinline__ void qkvs_mat(
    const __bf16 (&sA)[128][136], const __bf16* __restrict__ Wt,
    const float* __restrict__ bb, float* __restrict__ oo,
    int ar, int akOff, int bkOff, int rowTop, int lane) {
  v8f acc[8];
#pragma unroll
  for (int i = 0; i < 8; ++i)
#pragma unroll
    for (int j = 0; j < 8; ++j) acc[i][j] = 0.f;

#pragma unroll
  for (int k0 = 0; k0 < 128; k0 += 32) {
    Frag a;
    a.u[0] = *(const u32x4*)&sA[ar][k0 + akOff];
    a.u[1] = *(const u32x4*)&sA[ar][k0 + akOff + 16];
#pragma unroll
    for (int nt = 0; nt < 8; ++nt) {
      const __bf16* wr = Wt + (size_t)((nt << 4) + (lane & 15)) * 128 + k0 + bkOff;
      Frag b;
      b.u[0] = gload_u4(wr);
      b.u[1] = gload_u4(wr + 16);
      acc[nt] = __builtin_amdgcn_wmma_f32_16x16x32_bf16(
          false, a.v, false, b.v, (short)0, acc[nt], false, false);
    }
  }
#pragma unroll
  for (int nt = 0; nt < 8; ++nt) {
    int col = (nt << 4) + (lane & 15);
    float bv = bb[col];
#pragma unroll
    for (int j = 0; j < 8; ++j)
      gstore_f(oo + (size_t)(rowTop + j) * 128 + col, acc[nt][j] + bv);
  }
}

__global__ __launch_bounds__(256) void gemm_qkvs(
    const float* __restrict__ X,
    const __bf16* __restrict__ W0, const __bf16* __restrict__ W1,
    const __bf16* __restrict__ W2, const __bf16* __restrict__ W3,
    const float* __restrict__ b0, const float* __restrict__ b1,
    const float* __restrict__ b2, const float* __restrict__ b3,
    float* __restrict__ o0, float* __restrict__ o1,
    float* __restrict__ o2, float* __restrict__ o3) {
  __shared__ __align__(16) __bf16 sA[128][136];
  const int tid = threadIdx.x, lane = tid & 31, wave = tid >> 5;
  const int mBase = blockIdx.x * 128;
  {
    int r = tid >> 1, c0 = (tid & 1) * 64;
    const float* xp = X + (size_t)(mBase + r) * 128 + c0;
#pragma unroll
    for (int c = 0; c < 64; c += 8) {
      f32x4 f0 = gload_f4(xp + c), f1 = gload_f4(xp + c + 4);
      Pack8 p; pack8(p, f0, f1);
      *(u32x4*)&sA[r][c0 + c] = p.u;
    }
  }
  __syncthreads();

  int ar = (wave << 4) + (lane & 15);
  int akOff = (lane < 16) ? 0 : 8;
  int bkOff = (lane < 16) ? 0 : 8;
  int rowTop = mBase + (wave << 4) + ((lane < 16) ? 0 : 8);

  qkvs_mat(sA, W0, b0, o0, ar, akOff, bkOff, rowTop, lane);
  qkvs_mat(sA, W1, b1, o1, ar, akOff, bkOff, rowTop, lane);
  qkvs_mat(sA, W2, b2, o2, ar, akOff, bkOff, rowTop, lane);
  qkvs_mat(sA, W3, b3, o3, ar, akOff, bkOff, rowTop, lane);
}

// ---------------------------------------------------------------------------
// Fully fused TransformerConv attention, one block per graph (256 thr):
//   - stage q,k tiles (2 x 64KB LDS)
//   - per-edge scores, per-node softmax via ds_max_u32 / ds_add_f32
//   - message aggregation into LDS tile (reusing the q tile), + skip, flush.
// dyn LDS: 2 * 128*128 floats = 128KB.
// ---------------------------------------------------------------------------
__global__ __launch_bounds__(256) void k_att_fused(
    const float* __restrict__ q, const float* __restrict__ k,
    const float* __restrict__ v, const float* __restrict__ e,
    const int* __restrict__ src, const int* __restrict__ dst,
    const float* __restrict__ skip, float* __restrict__ out) {
  extern __shared__ float sm[];
  float* sQ = sm;                 // 128*128 floats (later reused as accumulator)
  float* sK = sm + PP * HIDn;     // 128*128 floats
  __shared__ float sAlpha[EPGn * 2];
  __shared__ unsigned sMaxU[PP * 2];
  __shared__ float sSum[PP * 2];
  __shared__ int sSrcL[EPGn], sDstL[EPGn];

  const int g = blockIdx.x, tid = threadIdx.x;
  const int nBase = g * PP, eBase = g * EPGn;
  const int lane = tid & 31, wv = tid >> 5;

  for (int i = tid; i < PP * HIDn; i += 256) {
    sQ[i] = q[(size_t)nBase * HIDn + i];
    sK[i] = k[(size_t)nBase * HIDn + i];
  }
  for (int i = tid; i < PP * 2; i += 256) { sMaxU[i] = 0u; sSum[i] = 0.f; }
  for (int i = tid; i < EPGn; i += 256) {
    sSrcL[i] = src[eBase + i] - nBase;
    sDstL[i] = dst[eBase + i] - nBase;
  }
  __syncthreads();

  // scores: one wave per edge
  for (int eb = 0; eb < EPGn; eb += 8) {
    int el = eb + wv;
    int sl = sSrcL[el], dl = sDstL[el];
    const float* ep = e + (size_t)(eBase + el) * HIDn;
    float p0 = sQ[dl * HIDn + lane] * (sK[sl * HIDn + lane] + ep[lane]) +
               sQ[dl * HIDn + lane + 32] * (sK[sl * HIDn + lane + 32] + ep[lane + 32]);
    float p1 = sQ[dl * HIDn + lane + 64] * (sK[sl * HIDn + lane + 64] + ep[lane + 64]) +
               sQ[dl * HIDn + lane + 96] * (sK[sl * HIDn + lane + 96] + ep[lane + 96]);
    for (int o = 16; o > 0; o >>= 1) {
      p0 += __shfl_down(p0, o, 32);
      p1 += __shfl_down(p1, o, 32);
    }
    if (lane == 0) {
      p0 *= 0.125f; p1 *= 0.125f;
      sAlpha[el * 2 + 0] = p0; sAlpha[el * 2 + 1] = p1;
      atomicMax(&sMaxU[dl * 2 + 0], fenc(p0));
      atomicMax(&sMaxU[dl * 2 + 1], fenc(p1));
    }
  }
  __syncthreads();
  // exp + segment sums
  for (int i = tid; i < EPGn * 2; i += 256) {
    int el = i >> 1, h = i & 1, dl = sDstL[el];
    float wv2 = __expf(sAlpha[i] - fdec(sMaxU[dl * 2 + h]));
    sAlpha[i] = wv2;
    atomicAdd(&sSum[dl * 2 + h], wv2);
  }
  __syncthreads();
  for (int i = tid; i < EPGn * 2; i += 256) {
    int el = i >> 1, h = i & 1, dl = sDstL[el];
    sAlpha[i] = sAlpha[i] / sSum[dl * 2 + h];
  }
  __syncthreads();
  // reuse sQ as the output accumulator tile
  float* acc = sQ;
  for (int i = tid; i < PP * HIDn; i += 256) acc[i] = 0.f;
  __syncthreads();
  for (int i = tid; i < EPGn * HIDn; i += 256) {
    int el = i >> 7, f = i & (HIDn - 1);
    int sl = sSrcL[el], dl = sDstL[el];
    float c = sAlpha[el * 2 + (f >> 6)];
    float contrib = (v[(size_t)(nBase + sl) * HIDn + f] +
                     e[(size_t)(eBase + el) * HIDn + f]) * c;
    atomicAdd(&acc[dl * HIDn + f], contrib);
  }
  __syncthreads();
  const float* sp = skip + (size_t)nBase * HIDn;
  float* op = out + (size_t)nBase * HIDn;
  for (int i = tid; i < PP * HIDn; i += 256) gstore_f(op + i, acc[i] + sp[i]);
}

// ---------------------------------------------------------------------------
// PyG graph-mode LayerNorm over each graph's 128x128 block (+ELU).
// ---------------------------------------------------------------------------
__global__ __launch_bounds__(256) void k_graph_norm_elu(
    const float* __restrict__ xin, float* __restrict__ xout,
    const float* __restrict__ w, const float* __restrict__ b) {
  const int g = blockIdx.x, tid = threadIdx.x;
  const float* xp = xin + (size_t)g * PP * HIDn;
  float s = 0.f, s2 = 0.f;
  for (int i = tid; i < PP * HIDn; i += 256) { float v = xp[i]; s += v; s2 += v * v; }
  __shared__ float rs[256], rs2[256];
  rs[tid] = s; rs2[tid] = s2; __syncthreads();
  for (int o = 128; o > 0; o >>= 1) {
    if (tid < o) { rs[tid] += rs[tid + o]; rs2[tid] += rs2[tid + o]; }
    __syncthreads();
  }
  float mean = rs[0] * (1.f / (PP * HIDn));
  float var  = rs2[0] * (1.f / (PP * HIDn)) - mean * mean;
  float rstd = rsqrtf(fmaxf(var, 0.f) + 1e-5f);
  float* op = xout + (size_t)g * PP * HIDn;
  for (int i = tid; i < PP * HIDn; i += 256) {
    int f = i & (HIDn - 1);
    gstore_f(op + i, elu1((xp[i] - mean) * rstd * w[f] + b[f]));
  }
}

__global__ void k_elu_inplace(float* __restrict__ p, int n) {
  int i = blockIdx.x * 256 + threadIdx.x;
  if (i < n) p[i] = elu1(p[i]);
}

// ---------------------------------------------------------------------------
// SAGPooling softmax + global_add_pool + edge pooling, per graph.
// ---------------------------------------------------------------------------
__global__ __launch_bounds__(256) void k_pool(
    const float* __restrict__ x, const float* __restrict__ ea,
    const int* __restrict__ src, const int* __restrict__ dst,
    const float* __restrict__ Wrel, const float* __restrict__ brel,
    const float* __restrict__ Wroot, const float* __restrict__ reW,
    const float* __restrict__ reB, float* __restrict__ gout) {
  __shared__ float sT[PP], sScore[PP], sG[HIDn], sEa[EFn], sRed[256];
  int g = blockIdx.x, tid = threadIdx.x;
  const float* xp = x + (size_t)g * PP * HIDn;

  {  // t[n] = <x[n],Wrel>, score[n] = <x[n],Wroot> + brel
    int n = tid >> 1, h = tid & 1;
    const float* row = xp + n * HIDn + h * 64;
    float a = 0.f, b = 0.f;
    for (int i = 0; i < 64; ++i) { float xv = row[i]; a += xv * Wrel[h * 64 + i]; b += xv * Wroot[h * 64 + i]; }
    sRed[tid] = a; __syncthreads();
    if (h == 0) sT[n] = sRed[tid] + sRed[tid + 1];
    __syncthreads();
    sRed[tid] = b; __syncthreads();
    if (h == 0) sScore[n] = sRed[tid] + sRed[tid + 1] + brel[0];
    __syncthreads();
  }
  {  // scatter: score[dst] += t[src]
    int eg = g * EPGn + tid;
    int sl = src[eg] - g * PP, dl = dst[eg] - g * PP;
    atomicAdd(&sScore[dl], sT[sl]);
    __syncthreads();
  }
  {  // softmax over 128 nodes
    float vv = (tid < PP) ? sScore[tid] : -1e30f;
    sRed[tid] = vv; __syncthreads();
    for (int o = 128; o > 0; o >>= 1) { if (tid < o) sRed[tid] = fmaxf(sRed[tid], sRed[tid + o]); __syncthreads(); }
    float mx = sRed[0]; __syncthreads();
    float ex = (tid < PP) ? __expf(sScore[tid] - mx) : 0.f;
    sRed[tid] = ex; __syncthreads();
    for (int o = 128; o > 0; o >>= 1) { if (tid < o) sRed[tid] += sRed[tid + o]; __syncthreads(); }
    float z = sRed[0]; __syncthreads();
    if (tid < PP) sScore[tid] = ex / z;
    __syncthreads();
  }
  {  // g[f] = sum_n x[n,f] * score[n]
    int f = tid & (HIDn - 1), h = tid >> 7;
    float a = 0.f;
    for (int n = h * 64; n < h * 64 + 64; ++n) a += xp[n * HIDn + f] * sScore[n];
    sRed[tid] = a; __syncthreads();
    if (h == 0) sG[f] = sRed[f] + sRed[f + 128];
    __syncthreads();
  }
  {  // sEa[f] = sum over 256 edges of ea[e,f]
    int f = tid & 63, seg = tid >> 6;
    const float* ep = ea + (size_t)g * EPGn * EFn;
    float a = 0.f;
    for (int e2 = seg * 64; e2 < seg * 64 + 64; ++e2) a += ep[e2 * EFn + f];
    sRed[tid] = a; __syncthreads();
    if (seg == 0) sEa[f] = sRed[f] + sRed[f + 64] + sRed[f + 128] + sRed[f + 192];
    __syncthreads();
  }
  if (tid < HIDn) {  // out = g * elu(sEa @ reW + reB)
    float a = reB[tid];
    for (int c = 0; c < EFn; ++c) a += sEa[c] * reW[c * HIDn + tid];
    gout[(size_t)g * HIDn + tid] = sG[tid] * elu1(a);
  }
}

// heads[b,i,:] = l2norm(repr[i,b,:] + finp[b,:])
__global__ __launch_bounds__(128) void k_heads(
    const float* __restrict__ repr, const float* __restrict__ finp,
    float* __restrict__ heads) {
  int b = blockIdx.x >> 2, i = blockIdx.x & 3, f = threadIdx.x;
  float v = repr[((size_t)i * BB + b) * HIDn + f] + finp[(size_t)b * HIDn + f];
  __shared__ float rd[128];
  rd[f] = v * v; __syncthreads();
  for (int o = 64; o > 0; o >>= 1) { if (f < o) rd[f] += rd[f + o]; __syncthreads(); }
  float sc = 1.f / fmaxf(sqrtf(rd[0]), 1e-12f);
  heads[((size_t)b * NBLKn + i) * HIDn + f] = v * sc;
}

// ---------------------------------------------------------------------------
// CoAttention + RESCAL per graph pair. dyn LDS: 128*129 floats (rel matrix).
// ---------------------------------------------------------------------------
__global__ __launch_bounds__(256) void k_final(
    const float* __restrict__ heads, const float* __restrict__ tails,
    const float* __restrict__ wq, const float* __restrict__ wk,
    const float* __restrict__ bias, const float* __restrict__ av,
    const float* __restrict__ relEmb, const int* __restrict__ rels,
    float* __restrict__ outp) {
  extern __shared__ float sR[];  // 128 x 129 (padded)
  __shared__ float sH[4 * 128], sTl[4 * 128], sK[4 * 64], sQ[4 * 64];
  __shared__ float sAtt[16], sM[4 * 128], sRed[256];
  int b = blockIdx.x, tid = threadIdx.x;
  for (int i = tid; i < 512; i += 256) {
    sH[i]  = heads[(size_t)b * 512 + i];
    sTl[i] = tails[(size_t)b * 512 + i];
  }
  const float* rr = relEmb + (size_t)rels[b] * (128 * 128);
  float ss = 0.f;
  for (int i = tid; i < 128 * 128; i += 256) {
    float v = rr[i]; ss += v * v;
    sR[(i >> 7) * 129 + (i & 127)] = v;
  }
  sRed[tid] = ss; __syncthreads();
  for (int o = 128; o > 0; o >>= 1) { if (tid < o) sRed[tid] += sRed[tid + o]; __syncthreads(); }
  float scale = 1.f / fmaxf(sqrtf(sRed[0]), 1e-12f);
  __syncthreads();
  {
    int i = tid >> 6, c = tid & 63;
    float a = 0.f, q2 = 0.f;
    for (int l = 0; l < 128; ++l) {
      a  += sH[i * 128 + l]  * wk[l * 64 + c];
      q2 += sTl[i * 128 + l] * wq[l * 64 + c];
    }
    sK[i * 64 + c] = a; sQ[i * 64 + c] = q2;
  }
  __syncthreads();
  if (tid < 16) {
    int i = tid >> 2, j = tid & 3;
    float a = 0.f;
    for (int c = 0; c < 64; ++c) a += tanhf(sK[i * 64 + c] + sQ[j * 64 + c] + bias[c]) * av[c];
    sAtt[tid] = a;
  }
  for (int oidx = tid; oidx < 512; oidx += 256) {
    int i = oidx >> 7, kc = oidx & 127;
    float a = 0.f;
    for (int l = 0; l < 128; ++l) a += sH[i * 128 + l] * sR[l * 129 + kc];
    sM[oidx] = a * scale;
  }
  __syncthreads();
  float part = 0.f;
  if (tid < 16) {
    int i = tid >> 2, j = tid & 3;
    float a = 0.f;
    for (int kk = 0; kk < 128; ++kk) a += sM[i * 128 + kk] * sTl[j * 128 + kk];
    part = sAtt[tid] * a;
  }
  sRed[tid] = part; __syncthreads();
  for (int o = 128; o > 0; o >>= 1) { if (tid < o) sRed[tid] += sRed[tid + o]; __syncthreads(); }
  if (tid == 0) outp[b] = sRed[0];
}

// ---------------------------------------------------------------------------
// Host side
// ---------------------------------------------------------------------------
static void gemmT(const float* X, const __bf16* Wt, const float* bias, float* out,
                  int M, int N, int K, int Kp, int act, hipStream_t st) {
  dim3 grid(M / 128), blk(256);
  if (N == 128) {
    if (act == 0)      gemm_wmma<0, 8><<<grid, blk, 0, st>>>(X, Wt, bias, out, M, N, K, Kp);
    else if (act == 1) gemm_wmma<1, 8><<<grid, blk, 0, st>>>(X, Wt, bias, out, M, N, K, Kp);
    else               gemm_wmma<2, 8><<<grid, blk, 0, st>>>(X, Wt, bias, out, M, N, K, Kp);
  } else {
    if (act == 0)      gemm_wmma<0, 4><<<grid, blk, 0, st>>>(X, Wt, bias, out, M, N, K, Kp);
    else if (act == 1) gemm_wmma<1, 4><<<grid, blk, 0, st>>>(X, Wt, bias, out, M, N, K, Kp);
    else               gemm_wmma<2, 4><<<grid, blk, 0, st>>>(X, Wt, bias, out, M, N, K, Kp);
  }
}

extern "C" void kernel_launch(void* const* d_in, const int* in_sizes, int n_in,
                              void* d_out, int out_size, void* d_ws, size_t ws_size,
                              hipStream_t stream) {
  (void)in_sizes; (void)n_in; (void)out_size; (void)ws_size;
  auto F = [&](int i) { return (const float*)d_in[i]; };

  const float* h_x  = F(0);
  const int*   h_src = (const int*)d_in[1];
  const int*   h_dst = h_src + EE;
  const float* h_ea = F(2);
  const float* t_x  = F(5);
  const int*   t_src = (const int*)d_in[6];
  const int*   t_dst = t_src + EE;
  const float* t_ea = F(7);
  const float* h_fin = F(10);
  const float* t_fin = F(12);
  const int*   rels  = (const int*)d_in[14];

  // params (setup_inputs dict insertion order, flattened)
  const float* init_node_W = F(15); const float* init_node_b = F(16);
  const float* init_edge_W = F(17); const float* init_edge_b = F(18);
  const float* init_norm_w = F(19); const float* init_norm_b = F(20);
  const float* cWq[2]  = {F(21), F(30)};
  const float* cWk[2]  = {F(22), F(31)};
  const float* cWv[2]  = {F(23), F(32)};
  const float* cWsk[2] = {F(24), F(33)};
  const float* cbq[2]  = {F(25), F(34)};
  const float* cbk[2]  = {F(26), F(35)};
  const float* cbv[2]  = {F(27), F(36)};
  const float* cbsk[2] = {F(28), F(37)};
  const float* cWe[2]  = {F(29), F(38)};
  const float* up1W = F(39); const float* up1b = F(40);
  const float* up2W = F(41); const float* up2b = F(42);
  const float* sagWrel = F(43); const float* sagBrel = F(44); const float* sagWroot = F(45);
  const float* n1w = F(46); const float* n1b = F(47);
  const float* n2w = F(48); const float* n2b = F(49);
  const float* reW = F(50); const float* reB = F(51);
  const float* fpW = F(52); const float* fpb = F(53);
  const float* attWq = F(56); const float* attWk = F(57);
  const float* attBias = F(58); const float* attA = F(59);
  const float* relEmb = F(60);

  // workspace layout (floats)
  float* w = (float*)d_ws; size_t o = 0;
  auto alloc = [&](size_t n) { float* p = w + o; o += n; return p; };
  float* xA    = alloc((size_t)NN * HIDn);
  float* xB    = alloc((size_t)NN * HIDn);
  float* qb    = alloc((size_t)NN * HIDn);
  float* kb    = alloc((size_t)NN * HIDn);
  float* vb    = alloc((size_t)NN * HIDn);
  float* skipb = alloc((size_t)NN * HIDn);
  float* attn  = alloc((size_t)NN * HIDn);
  float* ebuf  = alloc((size_t)EE * HIDn);
  float* eaA   = alloc((size_t)EE * EFn);
  float* eaB   = alloc((size_t)EE * EFn);
  float* reprH = alloc((size_t)NBLKn * BB * HIDn);
  float* reprT = alloc((size_t)NBLKn * BB * HIDn);
  float* finpH = alloc((size_t)BB * HIDn);
  float* finpT = alloc((size_t)BB * HIDn);
  float* headsB = alloc((size_t)BB * NBLKn * HIDn);
  float* tailsB = alloc((size_t)BB * NBLKn * HIDn);

  // bf16 transposed weight pool
  auto makeWt = [&](const float* W, int K, int N, int batch) {
    int Kp = (K + 31) & ~31;
    size_t total = (size_t)batch * N * Kp;
    __bf16* dst = (__bf16*)(w + o); o += total / 2;
    k_wt<<<(int)((total + 255) / 256), 256, 0, stream>>>(W, dst, K, N, Kp, (int)total);
    return dst;
  };
  const __bf16* wtNode = makeWt(init_node_W, FNODE, HIDn, 1);   // Kp=96
  const __bf16* wtEdge = makeWt(init_edge_W, FEDGE, EFn, 1);    // Kp=32
  const __bf16* wtQ[2]  = {makeWt(cWq[0], HIDn, HIDn, 4),  makeWt(cWq[1], HIDn, HIDn, 4)};
  const __bf16* wtK[2]  = {makeWt(cWk[0], HIDn, HIDn, 4),  makeWt(cWk[1], HIDn, HIDn, 4)};
  const __bf16* wtV[2]  = {makeWt(cWv[0], HIDn, HIDn, 4),  makeWt(cWv[1], HIDn, HIDn, 4)};
  const __bf16* wtS[2]  = {makeWt(cWsk[0], HIDn, HIDn, 4), makeWt(cWsk[1], HIDn, HIDn, 4)};
  const __bf16* wtWe[2] = {makeWt(cWe[0], EFn, HIDn, 4),   makeWt(cWe[1], EFn, HIDn, 4)};
  const __bf16* wtUp1 = makeWt(up1W, EFn, EFn, 4);
  const __bf16* wtUp2 = makeWt(up2W, EFn, EFn, 4);
  const __bf16* wtFp  = makeWt(fpW, FPn, HIDn, 1);

  const size_t attLds   = (size_t)(2 * PP * HIDn) * sizeof(float);   // 128 KB
  const size_t finalLds = (size_t)(128 * 129) * sizeof(float);

  auto runSide = [&](const float* x_in, const float* ea_in, const int* src,
                     const int* dst, float* repr) {
    float* cur = xA; float* oth = xB;
    gemmT(x_in, wtNode, init_node_b, cur, NN, HIDn, FNODE, 96, 0, stream);
    k_graph_norm_elu<<<BB, 256, 0, stream>>>(cur, oth, init_norm_w, init_norm_b);
    { float* t = cur; cur = oth; oth = t; }
    gemmT(ea_in, wtEdge, init_edge_b, eaA, EE, EFn, FEDGE, 32, 1, stream);

    for (int i = 0; i < NBLKn; ++i) {
      auto conv = [&](int c, const float* eaPtr) {
        gemm_qkvs<<<NN / 128, 256, 0, stream>>>(
            cur, wtQ[c] + (size_t)i * 16384, wtK[c] + (size_t)i * 16384,
            wtV[c] + (size_t)i * 16384, wtS[c] + (size_t)i * 16384,
            cbq[c] + i * HIDn, cbk[c] + i * HIDn, cbv[c] + i * HIDn,
            cbsk[c] + i * HIDn, qb, kb, vb, skipb);
        gemmT(eaPtr, wtWe[c] + (size_t)i * 8192, nullptr, ebuf, EE, HIDn, EFn, 64, 0, stream);
        k_att_fused<<<BB, 256, attLds, stream>>>(qb, kb, vb, ebuf, src, dst, skipb, attn);
      };
      conv(0, eaA);                                                   // conv1
      k_graph_norm_elu<<<BB, 256, 0, stream>>>(attn, oth, n1w + i * HIDn, n1b + i * HIDn);
      { float* t = cur; cur = oth; oth = t; }
      gemmT(eaA, wtUp1 + (size_t)i * 4096, up1b + i * EFn, eaB, EE, EFn, EFn, 64, 1, stream);
      conv(1, eaB);                                                   // conv2 (x2 in attn)
      gemmT(eaB, wtUp2 + (size_t)i * 4096, up2b + i * EFn, eaA, EE, EFn, EFn, 64, 0, stream);
      k_pool<<<BB, 256, 0, stream>>>(attn, eaA, src, dst,
                                     sagWrel + i * HIDn, sagBrel + i, sagWroot + i * HIDn,
                                     reW + (size_t)i * EFn * HIDn, reB + i * HIDn,
                                     repr + (size_t)i * BB * HIDn);
      k_graph_norm_elu<<<BB, 256, 0, stream>>>(attn, oth, n2w + i * HIDn, n2b + i * HIDn);
      { float* t = cur; cur = oth; oth = t; }
      k_elu_inplace<<<(EE * EFn + 255) / 256, 256, 0, stream>>>(eaA, EE * EFn);
    }
  };

  runSide(h_x, h_ea, h_src, h_dst, reprH);
  runSide(t_x, t_ea, t_src, t_dst, reprT);

  gemmT(h_fin, wtFp, fpb, finpH, BB, HIDn, FPn, 1024, 2, stream);
  gemmT(t_fin, wtFp, fpb, finpT, BB, HIDn, FPn, 1024, 2, stream);
  k_heads<<<BB * NBLKn, 128, 0, stream>>>(reprH, finpH, headsB);
  k_heads<<<BB * NBLKn, 128, 0, stream>>>(reprT, finpT, tailsB);
  k_final<<<BB, 256, finalLds, stream>>>(headsB, tailsB, attWq, attWk, attBias,
                                         attA, relEmb, rels, (float*)d_out);
}
